// GroupQueryAtt1_40097814675881
// MI455X (gfx1250) — matrix operand order repaired and checked by reading, not compiled, not run on hardware
//
#include <hip/hip_runtime.h>
#include <hip/hip_fp16.h>

#define HIDDEN 2048
#define HEADS  16
#define NG     4          // kv groups
#define HPG    4          // heads per group
#define DH     128        // head dim
#define SEQ    2048
#define KVW    (NG * DH)  // 512

typedef __attribute__((ext_vector_type(16))) _Float16 v16h;
typedef __attribute__((ext_vector_type(8)))  _Float16 half8;
typedef __attribute__((ext_vector_type(4)))  _Float16 hx4;
typedef __attribute__((ext_vector_type(4)))  float    fx4;
typedef __attribute__((ext_vector_type(8)))  float    v8f;

// Build a WMMA 16-half fragment from two contiguous 8-half runs (2x ds_load_b128).
__device__ __forceinline__ v16h ld_frag(const _Float16* p0, const _Float16* p1) {
  half8 lo = *(const half8*)p0;
  half8 hi = *(const half8*)p1;
  v16h r;
#pragma unroll
  for (int i = 0; i < 8; ++i) { r[i] = lo[i]; r[8 + i] = hi[i]; }
  return r;
}

__device__ __forceinline__ float redmax16(float v) {
  v = fmaxf(v, __shfl_xor(v, 1, 32));
  v = fmaxf(v, __shfl_xor(v, 2, 32));
  v = fmaxf(v, __shfl_xor(v, 4, 32));
  v = fmaxf(v, __shfl_xor(v, 8, 32));
  return v;
}
__device__ __forceinline__ float redsum16(float v) {
  v += __shfl_xor(v, 1, 32);
  v += __shfl_xor(v, 2, 32);
  v += __shfl_xor(v, 4, 32);
  v += __shfl_xor(v, 8, 32);
  return v;
}

// Vector-load 4 elements (b128 for f32, b64 for f16) and expose as floats.
__device__ __forceinline__ void fetch4(const float* p, float v[4]) {
  fx4 t = *(const fx4*)p;
#pragma unroll
  for (int i = 0; i < 4; ++i) v[i] = t[i];
}
__device__ __forceinline__ void fetch4(const _Float16* p, float v[4]) {
  hx4 t = *(const hx4*)p;
#pragma unroll
  for (int i = 0; i < 4; ++i) v[i] = (float)t[i];
}

// C[M,N] = A[M,K] @ W[K,N] + bias, via v_wmma_f32_16x16x32_f16.
// Tile 64x128, 128 threads = 4 waves; each wave computes 64x32 of output
// (8 accumulator fragments -> 8 WMMA per 32-deep k-step).
template <typename AT, typename OT>
__global__ __launch_bounds__(128)
void gemm_bias_wmma(const AT* __restrict__ A, const float* __restrict__ W,
                    const float* __restrict__ bias, OT* __restrict__ C,
                    int M, int N, int K) {
  __shared__ _Float16 As[64][40];    // row-major A tile  (64 x 32), padded
  __shared__ _Float16 Bs[128][40];   // transposed W tile (128n x 32k), padded

  const int t    = threadIdx.x;
  const int lane = t & 31;
  const int w    = t >> 5;
  const int h    = lane >> 4;   // half-wave
  const int ln   = lane & 15;

  const int m0 = blockIdx.y * 64;
  const int n0 = blockIdx.x * 128;
  const int wn = w * 32;        // wave's 32-wide column strip

  v8f acc[4][2];
#pragma unroll
  for (int mt = 0; mt < 4; ++mt) { acc[mt][0] = (v8f){}; acc[mt][1] = (v8f){}; }

  for (int k0 = 0; k0 < K; k0 += 32) {
    // Prefetch next k-tile into L2 (global_prefetch_b8); whole set lives in 192MB L2.
    if (k0 + 32 < K) {
      __builtin_prefetch(A + (size_t)(m0 + (t >> 3)) * K + (k0 + 32) + (t & 7) * 4, 0, 3);
      __builtin_prefetch(W + (size_t)(k0 + 32 + (t >> 5)) * N + n0 + (t & 31) * 4, 0, 3);
    }
    __syncthreads();
    // A tile 64x32: 512 x 4-elem vectors, b128/b64 global loads, b64 LDS stores
#pragma unroll
    for (int i = 0; i < 4; ++i) {
      int s = i * 128 + t;
      int r = s >> 3, c4 = (s & 7) * 4;
      float v[4];
      fetch4(A + (size_t)(m0 + r) * K + k0 + c4, v);
      hx4 hv;
#pragma unroll
      for (int jj = 0; jj < 4; ++jj) hv[jj] = (_Float16)v[jj];
      *(hx4*)&As[r][c4] = hv;
    }
    // W tile 32x128: b128 loads coalesced along N, transposed scalar stores
#pragma unroll
    for (int i = 0; i < 8; ++i) {
      int s = i * 128 + t;
      int kk = s >> 5, n4 = (s & 31) * 4;
      float v[4];
      fetch4(W + (size_t)(k0 + kk) * N + n0 + n4, v);
#pragma unroll
      for (int jj = 0; jj < 4; ++jj) Bs[n4 + jj][kk] = (_Float16)v[jj];
    }
    __syncthreads();

    // B frags: lane col = wn+ln(+16), halves K = 8h..8h+7 and 16+8h..16+8h+7
    v16h b0 = ld_frag(&Bs[wn + ln][8 * h],      &Bs[wn + ln][16 + 8 * h]);
    v16h b1 = ld_frag(&Bs[wn + 16 + ln][8 * h], &Bs[wn + 16 + ln][16 + 8 * h]);
#pragma unroll
    for (int mt = 0; mt < 4; ++mt) {
      // A frag: lane row = mt*16+ln, halves K = 8h..8h+7 and 16+8h..16+8h+7
      v16h a = ld_frag(&As[mt * 16 + ln][8 * h], &As[mt * 16 + ln][16 + 8 * h]);
      acc[mt][0] = __builtin_amdgcn_wmma_f32_16x16x32_f16(false, a, false, b0, (short)0, acc[mt][0], false, false);
      acc[mt][1] = __builtin_amdgcn_wmma_f32_16x16x32_f16(false, a, false, b1, (short)0, acc[mt][1], false, false);
    }
  }

  const float bb0 = bias[n0 + wn + ln];
  const float bb1 = bias[n0 + wn + 16 + ln];
#pragma unroll
  for (int mt = 0; mt < 4; ++mt) {
#pragma unroll
    for (int r = 0; r < 8; ++r) {
      int row = m0 + mt * 16 + r + 8 * h;
      C[(size_t)row * N + n0 + wn + ln]      = (OT)(acc[mt][0][r] + bb0);
      C[(size_t)row * N + n0 + wn + 16 + ln] = (OT)(acc[mt][1][r] + bb1);
    }
  }
}

// Flash-attention GQA: grid (SEQ/64, HEADS, B), 128 threads.
// Each wave handles 16 query rows of one head; K/V streamed in 32-key chunks.
__global__ __launch_bounds__(128)
void gqa_flash_wmma(const _Float16* __restrict__ qb, const _Float16* __restrict__ kb,
                    const _Float16* __restrict__ vb, _Float16* __restrict__ ob) {
  __shared__ _Float16 Kt[32][144];     // K chunk row-major [key][d]   (= B operand of S=Q.K^T)
  __shared__ _Float16 Vt[128][40];     // V chunk transposed [d][key]  (= B operand of O=P.V)
  __shared__ _Float16 Ps[4][16][40];   // per-wave P relayout scratch

  const int t    = threadIdx.x;
  const int lane = t & 31;
  const int w    = t >> 5;
  const int h    = lane >> 4;
  const int ln   = lane & 15;

  const int b    = blockIdx.z;
  const int head = blockIdx.y;
  const int g    = head >> 2;                 // head / HPG
  const int q0   = blockIdx.x * 64 + w * 16;  // wave's query tile base

  const float scale = 0.08838834764831845f;   // 1/sqrt(128)

  // Q fragments (16 rows x 128 d) live in registers for the whole kernel.
  v16h qa[4];
  {
    const _Float16* qrow = qb + (size_t)(b * SEQ + q0 + ln) * HIDDEN + head * DH;
#pragma unroll
    for (int kk = 0; kk < 4; ++kk) {
      half8 lo = *(const half8*)(qrow + kk * 32 + 8 * h);
      half8 hi = *(const half8*)(qrow + kk * 32 + 16 + 8 * h);
#pragma unroll
      for (int i = 0; i < 8; ++i) { qa[kk][i] = lo[i]; qa[kk][8 + i] = hi[i]; }
    }
  }

  v8f acc[8];
#pragma unroll
  for (int f = 0; f < 8; ++f) acc[f] = (v8f){};
  float mrow[8], lrow[8];
#pragma unroll
  for (int r = 0; r < 8; ++r) { mrow[r] = -1e30f; lrow[r] = 0.0f; }

  for (int j = 0; j < SEQ; j += 32) {
    {
      // Prefetch next chunk's K/V cachelines while we work on this one.
      int key = t >> 2, d8 = t & 3;   // 32 keys x 4 x 64B
      if (j + 32 < SEQ) {
        const size_t pb = (size_t)(b * SEQ + j + 32 + key) * KVW + g * DH + d8 * 32;
        __builtin_prefetch(kb + pb, 0, 3);
        __builtin_prefetch(vb + pb, 0, 3);
      }
    }
    __syncthreads();
    // Cooperative load: K chunk (row-major copy) + V chunk (transposed)
#pragma unroll
    for (int i = 0; i < 4; ++i) {
      int slot = i * 128 + t;             // 512 slots of 8 halfs
      int key = slot >> 4, d8 = slot & 15;
      const size_t base = (size_t)(b * SEQ + j + key) * KVW + g * DH + d8 * 8;
      half8 kv = *(const half8*)(kb + base);
      *(half8*)&Kt[key][d8 * 8] = kv;
      half8 vv = *(const half8*)(vb + base);
#pragma unroll
      for (int q = 0; q < 8; ++q) Vt[d8 * 8 + q][key] = vv[q];
    }
    __syncthreads();

    // Score tiles: s0 = keys j..j+15, s1 = keys j+16..j+31
    v8f s0 = {}, s1 = {};
#pragma unroll
    for (int kk = 0; kk < 4; ++kk) {
      v16h kb0 = ld_frag(&Kt[ln][kk * 32 + 8 * h],      &Kt[ln][kk * 32 + 16 + 8 * h]);
      v16h kb1 = ld_frag(&Kt[16 + ln][kk * 32 + 8 * h], &Kt[16 + ln][kk * 32 + 16 + 8 * h]);
      s0 = __builtin_amdgcn_wmma_f32_16x16x32_f16(false, qa[kk], false, kb0, (short)0, s0, false, false);
      s1 = __builtin_amdgcn_wmma_f32_16x16x32_f16(false, qa[kk], false, kb1, (short)0, s1, false, false);
    }

    // Online softmax: per row (r + 8h), values spread across the 16-lane group
    float alpha[8];
#pragma unroll
    for (int r = 0; r < 8; ++r) {
      float a  = s0[r] * scale;
      float bb = s1[r] * scale;
      float mx = redmax16(fmaxf(a, bb));
      float mn = fmaxf(mrow[r], mx);
      float p0 = __expf(a - mn);
      float p1 = __expf(bb - mn);
      alpha[r] = __expf(mrow[r] - mn);
      mrow[r]  = mn;
      lrow[r]  = lrow[r] * alpha[r] + redsum16(p0 + p1);
      Ps[w][r + 8 * h][ln]      = (_Float16)p0;   // C-layout -> row-major scratch
      Ps[w][r + 8 * h][16 + ln] = (_Float16)p1;
    }
    // Intra-wave cross-lane LDS RAW: wait for the P stores before re-reading.
    asm volatile("s_wait_dscnt 0" ::: "memory");

    // Reload P in WMMA A-fragment layout (16 x 32)
    v16h pa = ld_frag(&Ps[w][ln][8 * h], &Ps[w][ln][16 + 8 * h]);

    // O += P @ V  (8 column fragments of 16 d each), with accumulator rescale
#pragma unroll
    for (int f = 0; f < 8; ++f) {
#pragma unroll
      for (int r = 0; r < 8; ++r) acc[f][r] *= alpha[r];
      v16h vfrag = ld_frag(&Vt[f * 16 + ln][8 * h], &Vt[f * 16 + ln][16 + 8 * h]);
      acc[f] = __builtin_amdgcn_wmma_f32_16x16x32_f16(false, pa, false, vfrag, (short)0, acc[f], false, false);
    }
  }

  // Normalize and store O (f16) in (B,S,HIDDEN) layout for the final GEMM
#pragma unroll
  for (int f = 0; f < 8; ++f) {
#pragma unroll
    for (int r = 0; r < 8; ++r) {
      int row = q0 + r + 8 * h;
      ob[(size_t)(b * SEQ + row) * HIDDEN + head * DH + f * 16 + ln] =
          (_Float16)(acc[f][r] / lrow[r]);
    }
  }
}

extern "C" void kernel_launch(void* const* d_in, const int* in_sizes, int n_in,
                              void* d_out, int out_size, void* d_ws, size_t ws_size,
                              hipStream_t stream) {
  const float* x  = (const float*)d_in[0];
  const float* Wq = (const float*)d_in[1];
  const float* bq = (const float*)d_in[2];
  const float* Wk = (const float*)d_in[3];
  const float* bk = (const float*)d_in[4];
  const float* Wv = (const float*)d_in[5];
  const float* bv = (const float*)d_in[6];
  const float* Wo = (const float*)d_in[7];
  const float* bo = (const float*)d_in[8];
  float* out = (float*)d_out;

  const int B = in_sizes[0] / (SEQ * HIDDEN);   // = 2
  const int M = B * SEQ;                        // = 4096

  // f16 workspace: q (16MB), k (4MB), v (4MB), o (16MB)
  _Float16* q_h = (_Float16*)d_ws;
  _Float16* k_h = q_h + (size_t)M * HIDDEN;
  _Float16* v_h = k_h + (size_t)M * KVW;
  _Float16* o_h = v_h + (size_t)M * KVW;

  dim3 blk(128);
  gemm_bias_wmma<float, _Float16>
      <<<dim3(HIDDEN / 128, M / 64), blk, 0, stream>>>(x, Wq, bq, q_h, M, HIDDEN, HIDDEN);
  gemm_bias_wmma<float, _Float16>
      <<<dim3(KVW / 128, M / 64), blk, 0, stream>>>(x, Wk, bk, k_h, M, KVW, HIDDEN);
  gemm_bias_wmma<float, _Float16>
      <<<dim3(KVW / 128, M / 64), blk, 0, stream>>>(x, Wv, bv, v_h, M, KVW, HIDDEN);

  gqa_flash_wmma<<<dim3(SEQ / 64, HEADS, B), blk, 0, stream>>>(q_h, k_h, v_h, o_h);

  gemm_bias_wmma<_Float16, float>
      <<<dim3(HIDDEN / 128, M / 64), blk, 0, stream>>>(o_h, Wo, bo, out, M, HIDDEN, HIDDEN);
}
